// EncoderLayer_39376260170324
// MI455X (gfx1250) — matrix operand order repaired and checked by reading, not compiled, not run on hardware
//
#include <hip/hip_runtime.h>

// ---------------------------------------------------------------------------
// BERT encoder layer for MI455X (gfx1250, wave32, WMMA).
// B=2, S=2048, D=1024, HID=4096, NHEAD=16, dh=64.
// bf16 WMMA (16x16x32), fp32 accumulate, flash-attention streaming softmax,
// async global->LDS (b128) tile staging, weights pre-transposed so all tile
// copies are K-contiguous 128-bit transfers.
// ---------------------------------------------------------------------------

typedef __attribute__((ext_vector_type(16))) __bf16 v16bf;
typedef __attribute__((ext_vector_type(8)))  float  v8f;
typedef int v4i_vec __attribute__((vector_size(16)));   // int4 vector for async builtin

union BFrag {           // 8 VGPRs = 16 bf16 halves (one WMMA A or B operand)
    uint4 q[2];
    v16bf v;
};

#ifndef __has_builtin
#define __has_builtin(x) 0
#endif
#if __has_builtin(__builtin_amdgcn_global_load_async_to_lds_b128)
#define ASYNC_COPY_BUILTIN 1
#else
#define ASYNC_COPY_BUILTIN 0
#endif
#if __has_builtin(__builtin_amdgcn_s_wait_asynccnt)
#define WAIT_ASYNC_BUILTIN 1
#else
#define WAIT_ASYNC_BUILTIN 0
#endif

// 16-byte global -> LDS async copy (ASYNCcnt-tracked, no VGPR round trip)
static __device__ __forceinline__ void copy16_g2l(const void* g, void* l) {
#if ASYNC_COPY_BUILTIN
    __builtin_amdgcn_global_load_async_to_lds_b128(
        (__attribute__((address_space(1))) v4i_vec*)(g),
        (__attribute__((address_space(3))) v4i_vec*)(l), 0, 0);
#else
    asm volatile("global_load_async_to_lds_b128 %0, %1, off"
                 :: "v"((unsigned)(unsigned long long)l), "v"(g)
                 : "memory");
#endif
}

static __device__ __forceinline__ void wait_async0() {
#if WAIT_ASYNC_BUILTIN
    __builtin_amdgcn_s_wait_asynccnt(0);
#else
    asm volatile("s_wait_asynccnt 0x0" ::: "memory");
#endif
}

static __device__ __forceinline__ unsigned short f2bf(float f) {
    unsigned u = __float_as_uint(f);
    unsigned r = u + 0x7fffu + ((u >> 16) & 1u);   // round-to-nearest-even
    return (unsigned short)(r >> 16);
}

// ---------------------------------------------------------------------------
// fp32 -> bf16 elementwise (activations)
// ---------------------------------------------------------------------------
__global__ __launch_bounds__(256) void cvt_f32_bf16(
    const float* __restrict__ in, unsigned short* __restrict__ out, int n) {
    int i = (blockIdx.x * 256 + threadIdx.x) * 4;
    if (i + 3 < n) {
        float4 f = *(const float4*)(in + i);
        union { uint2 q; unsigned short h[4]; } u;
        u.h[0] = f2bf(f.x); u.h[1] = f2bf(f.y);
        u.h[2] = f2bf(f.z); u.h[3] = f2bf(f.w);
        *(uint2*)(out + i) = u.q;
    }
}

// ---------------------------------------------------------------------------
// fp32 W[K,N] -> bf16 W^T[N,K] (tiled 32x32 transpose through LDS).
// One-time prep so GEMM B-tiles become K-contiguous b128 async copies.
// grid = (N/32, K/32), block = 256 (8x32)
// ---------------------------------------------------------------------------
__global__ __launch_bounds__(256) void cvt_transpose_bf16(
    const float* __restrict__ in, unsigned short* __restrict__ out,
    int K, int N) {
    __shared__ float t[32][33];
    const int n0 = blockIdx.x * 32, k0 = blockIdx.y * 32;
    const int tx = threadIdx.x & 31;
    const int ty = threadIdx.x >> 5;
    #pragma unroll
    for (int i = 0; i < 32; i += 8)
        t[ty + i][tx] = in[(size_t)(k0 + ty + i) * N + n0 + tx];
    __syncthreads();
    #pragma unroll
    for (int i = 0; i < 32; i += 8)
        out[(size_t)(n0 + ty + i) * K + k0 + tx] = f2bf(t[tx][ty + i]);
}

// ---------------------------------------------------------------------------
// Tiled bf16 GEMM:  C[M,N] = A[M,K] @ B[K,N] + bias, with B given as
// BT[N,K]. Block tile 128x64, BK=32, 8 waves; wave w -> 16 rows x 64 cols.
// Both LDS tiles ([row][k] and [n][k]) staged with async b128 copies.
// ISA bf16 A layout: lane<16 K0-7/K16-23, lane>=16 K8-15/K24-31.
// ISA bf16 B layout: lane<16 K0-7/K16-23, lane>=16 K8-15/K24-31, N = lane&15.
// ---------------------------------------------------------------------------
template <int RELU, int BF16OUT>
__global__ __launch_bounds__(256) void gemm_bf16(
    const unsigned short* __restrict__ A, const unsigned short* __restrict__ BT,
    const float* __restrict__ bias, void* __restrict__ Cout,
    int M, int N, int K) {
    __shared__ __align__(16) unsigned short ldsA[128 * 32];   // [row][k]
    __shared__ __align__(16) unsigned short ldsBT[64 * 32];   // [n][k]

    const int tid  = threadIdx.x;
    const int lane = tid & 31;
    const int w    = tid >> 5;
    const int m0   = blockIdx.y * 128;
    const int n0   = blockIdx.x * 64;

    v8f acc[4] = {{}, {}, {}, {}};

    for (int k0 = 0; k0 < K; k0 += 32) {
        __syncthreads();
        // --- async stage A tile 128x32 (2 x b128 per thread)
        #pragma unroll
        for (int it = 0; it < 2; ++it) {
            int idx = (tid + it * 256) * 8;
            int r = idx >> 5, c = idx & 31;
            copy16_g2l(A + (size_t)(m0 + r) * K + k0 + c, &ldsA[r * 32 + c]);
        }
        // --- async stage BT tile 64x32 (1 x b128 per thread)
        {
            int n = tid >> 2;
            int c = (tid & 3) * 8;
            copy16_g2l(BT + (size_t)(n0 + n) * K + k0 + c, &ldsBT[n * 32 + c]);
        }
        if (k0 + 32 < K)   // prefetch next A tile -> global_prefetch_b8
            __builtin_prefetch(A + (size_t)(m0 + (tid >> 1)) * K + k0 + 32, 0, 0);
        wait_async0();
        __syncthreads();

        // --- A fragment for this wave's 16 rows
        BFrag a;
        const int am = w * 16 + (lane & 15);
        const int g8 = (lane >> 4) * 8;
        a.q[0] = *(const uint4*)(&ldsA[am * 32 + g8]);
        a.q[1] = *(const uint4*)(&ldsA[am * 32 + 16 + g8]);

        const int kb = (lane >> 4) * 16;
        #pragma unroll
        for (int j = 0; j < 4; ++j) {
            BFrag b;
            const int bn = j * 16 + (lane & 15);
            b.q[0] = *(const uint4*)(&ldsBT[bn * 32 + g8]);
            b.q[1] = *(const uint4*)(&ldsBT[bn * 32 + 16 + g8]);
            acc[j] = __builtin_amdgcn_wmma_f32_16x16x32_bf16(
                false, a.v, false, b.v, (short)0, acc[j], false, false);
        }
    }

    // --- epilogue: bias (+ReLU), store bf16 or fp32
    const int mrow = m0 + w * 16 + (lane >> 4) * 8;
    const int ncol = n0 + (lane & 15);
    #pragma unroll
    for (int j = 0; j < 4; ++j) {
        int col = ncol + j * 16;
        float bv = bias[col];
        #pragma unroll
        for (int r = 0; r < 8; ++r) {
            float v = acc[j][r] + bv;
            if (RELU) v = fmaxf(v, 0.0f);
            if (BF16OUT)
                ((unsigned short*)Cout)[(size_t)(mrow + r) * N + col] = f2bf(v);
            else
                ((float*)Cout)[(size_t)(mrow + r) * N + col] = v;
        }
    }
}

// ---------------------------------------------------------------------------
// Flash attention. Grid: (S/128, B*NHEAD). 8 waves x 16 query rows.
// K tile (32x64) staged async; V tile transposed in LDS for the PV
// B-fragment; online softmax in the C/D layout with 16-lane butterflies.
// qkv layout: [B*S][3072] bf16, q=[0,1024), k=[1024,2048), v=[2048,3072)
// ---------------------------------------------------------------------------
__global__ __launch_bounds__(256) void attention_kernel(
    const unsigned short* __restrict__ qkv,
    const unsigned char* __restrict__ mask,
    unsigned short* __restrict__ attn_out) {
    __shared__ __align__(16) unsigned short ldsK[32 * 64];       // [key][dh]
    __shared__ __align__(16) unsigned short ldsVT[64 * 32];      // [dh][key]
    __shared__ __align__(16) unsigned short ldsP[8 * 16 * 32];   // per-wave P

    const int tid  = threadIdx.x;
    const int lane = tid & 31;
    const int w    = tid >> 5;
    const int bh   = blockIdx.y;          // b*16 + h
    const int b    = bh >> 4;
    const int h    = bh & 15;
    const int q0   = blockIdx.x * 128;
    const size_t rs3 = 3072;
    const size_t base = (size_t)b * 2048;

    const int m  = lane & 15;
    const int g8 = (lane >> 4) * 8;
    const int kb = (lane >> 4) * 16;

    // Q fragments: dh=64 -> two K=32 chunks
    BFrag qa[2];
    {
        const unsigned short* qrow = qkv + (base + q0 + w * 16 + m) * rs3 + h * 64;
        #pragma unroll
        for (int kc = 0; kc < 2; ++kc) {
            qa[kc].q[0] = *(const uint4*)(qrow + kc * 32 + g8);
            qa[kc].q[1] = *(const uint4*)(qrow + kc * 32 + 16 + g8);
        }
    }

    v8f ctx[4] = {{}, {}, {}, {}};
    float mrow[8], lrow[8];
    #pragma unroll
    for (int r = 0; r < 8; ++r) { mrow[r] = -1e30f; lrow[r] = 0.0f; }

    const float scale = 0.125f;   // 1/sqrt(64)

    for (int kt = 0; kt < 2048; kt += 32) {
        __syncthreads();
        // --- K tile: async b128 per thread; V tile: transpose through VALU
        {
            int kk = tid >> 3;
            int c  = (tid & 7) * 8;
            copy16_g2l(qkv + (base + kt + kk) * rs3 + 1024 + h * 64 + c,
                       &ldsK[kk * 64 + c]);
            union { uint4 q; unsigned short hx[8]; } u;
            u.q = *(const uint4*)(qkv + (base + kt + kk) * rs3 + 2048 + h * 64 + c);
            #pragma unroll
            for (int i = 0; i < 8; ++i) ldsVT[(c + i) * 32 + kk] = u.hx[i];
        }
        wait_async0();
        __syncthreads();

        // --- S = Q @ K^T : 16x32 in two 16x16 accumulators
        v8f s[2] = {{}, {}};
        #pragma unroll
        for (int j = 0; j < 2; ++j) {
            const int key = j * 16 + (lane & 15);
            #pragma unroll
            for (int kc = 0; kc < 2; ++kc) {
                BFrag bb;   // B[k=dh][n=key]: contiguous per lane in ldsK
                bb.q[0] = *(const uint4*)(&ldsK[key * 64 + kc * 32 + g8]);
                bb.q[1] = *(const uint4*)(&ldsK[key * 64 + kc * 32 + 16 + g8]);
                s[j] = __builtin_amdgcn_wmma_f32_16x16x32_bf16(
                    false, qa[kc].v, false, bb.v, (short)0, s[j], false, false);
            }
        }

        // --- scale + key mask
        float sc[2][8];
        #pragma unroll
        for (int j = 0; j < 2; ++j) {
            const int key = kt + j * 16 + (lane & 15);
            const bool mk = mask[base + key] != 0;
            #pragma unroll
            for (int r = 0; r < 8; ++r)
                sc[j][r] = mk ? -1e9f : s[j][r] * scale;
        }

        // --- row max over 32 cols (16-lane half-group butterfly)
        float mt[8];
        #pragma unroll
        for (int r = 0; r < 8; ++r) mt[r] = fmaxf(sc[0][r], sc[1][r]);
        #pragma unroll
        for (int d = 1; d < 16; d <<= 1)
            #pragma unroll
            for (int r = 0; r < 8; ++r)
                mt[r] = fmaxf(mt[r], __shfl_xor(mt[r], d, 32));

        // --- online softmax update
        float al[8], p0[8], p1[8], rsum[8];
        #pragma unroll
        for (int r = 0; r < 8; ++r) {
            float mn = fmaxf(mrow[r], mt[r]);
            al[r]   = __expf(mrow[r] - mn);
            p0[r]   = __expf(sc[0][r] - mn);
            p1[r]   = __expf(sc[1][r] - mn);
            rsum[r] = p0[r] + p1[r];
            mrow[r] = mn;
        }
        #pragma unroll
        for (int j = 0; j < 4; ++j)
            #pragma unroll
            for (int r = 0; r < 8; ++r) ctx[j][r] *= al[r];
        #pragma unroll
        for (int d = 1; d < 16; d <<= 1)
            #pragma unroll
            for (int r = 0; r < 8; ++r) rsum[r] += __shfl_xor(rsum[r], d, 32);
        #pragma unroll
        for (int r = 0; r < 8; ++r) lrow[r] = lrow[r] * al[r] + rsum[r];

        // --- P (C-layout) -> bf16 A-layout via per-wave LDS scratch
        unsigned short* pb = &ldsP[w * 512];
        {
            const int c  = lane & 15;
            const int rb = (lane >> 4) * 8;
            #pragma unroll
            for (int r = 0; r < 8; ++r) {
                pb[(rb + r) * 32 + c]      = f2bf(p0[r]);
                pb[(rb + r) * 32 + 16 + c] = f2bf(p1[r]);
            }
        }
        __syncthreads();

        // --- ctx += P(16x32) @ V(32x64)
        BFrag pa;
        pa.q[0] = *(const uint4*)(pb + m * 32 + g8);
        pa.q[1] = *(const uint4*)(pb + m * 32 + 16 + g8);
        #pragma unroll
        for (int j = 0; j < 4; ++j) {
            BFrag bb;
            const int n = j * 16 + (lane & 15);
            bb.q[0] = *(const uint4*)(&ldsVT[n * 32 + g8]);
            bb.q[1] = *(const uint4*)(&ldsVT[n * 32 + 16 + g8]);
            ctx[j] = __builtin_amdgcn_wmma_f32_16x16x32_bf16(
                false, pa.v, false, bb.v, (short)0, ctx[j], false, false);
        }
    }

    // --- normalize by row sum, store bf16 into [B*S][D] at head offset
    const size_t orow = base + q0 + w * 16 + (lane >> 4) * 8;
    #pragma unroll
    for (int j = 0; j < 4; ++j) {
        const int col = h * 64 + j * 16 + (lane & 15);
        #pragma unroll
        for (int r = 0; r < 8; ++r) {
            float v = ctx[j][r] * (1.0f / lrow[r]);
            attn_out[(orow + r) * 1024 + col] = f2bf(v);
        }
    }
}

// ---------------------------------------------------------------------------
// LayerNorm over D=1024, one block (8 waves) per token row.
// ---------------------------------------------------------------------------
__global__ __launch_bounds__(256) void layernorm_kernel(
    const float* __restrict__ X, const float* __restrict__ gamma,
    const float* __restrict__ beta, float* __restrict__ out) {
    const int row  = blockIdx.x;
    const int tid  = threadIdx.x;
    const int lane = tid & 31;
    const int w    = tid >> 5;
    const float* xr = X + (size_t)row * 1024;

    float v[4], s1 = 0.0f, s2 = 0.0f;
    #pragma unroll
    for (int i = 0; i < 4; ++i) {
        v[i] = xr[tid + i * 256];
        s1 += v[i];
        s2 += v[i] * v[i];
    }
    #pragma unroll
    for (int d = 1; d < 32; d <<= 1) {
        s1 += __shfl_xor(s1, d, 32);
        s2 += __shfl_xor(s2, d, 32);
    }
    __shared__ float red[2][8];
    if (lane == 0) { red[0][w] = s1; red[1][w] = s2; }
    __syncthreads();
    float t1 = 0.0f, t2 = 0.0f;
    #pragma unroll
    for (int i = 0; i < 8; ++i) { t1 += red[0][i]; t2 += red[1][i]; }
    const float mean = t1 * (1.0f / 1024.0f);
    const float var  = t2 * (1.0f / 1024.0f) - mean * mean;
    const float rstd = rsqrtf(var + 1e-5f);
    float* orow = out + (size_t)row * 1024;
    #pragma unroll
    for (int i = 0; i < 4; ++i) {
        int c = tid + i * 256;
        orow[c] = (v[i] - mean) * rstd * gamma[c] + beta[c];
    }
}

// ---------------------------------------------------------------------------
// Workspace layout (bytes):
//   Xbf     [4096,1024] bf16 @        0  ( 8 MB)
//   WqkvT   [3072,1024] bf16 @  8388608  ( 6 MB)
//   QKVbf   [4096,3072] bf16 @ 14680064  (24 MB)
//   W1T     [4096,1024] bf16 @ 39845888  ( 8 MB)
//   W2T     [1024,4096] bf16 @ 48234496  ( 8 MB)
//   Attnbf  [4096,1024] bf16 @ 56623104  ( 8 MB)
//   Hbf     [4096,4096] bf16 @ 65011712  (32 MB)
//   F2out   [4096,1024] f32  @ 98566144  (16 MB)   total ~110 MB
// ---------------------------------------------------------------------------
extern "C" void kernel_launch(void* const* d_in, const int* in_sizes, int n_in,
                              void* d_out, int out_size, void* d_ws, size_t ws_size,
                              hipStream_t stream) {
    const float* emb          = (const float*)d_in[0];
    const unsigned char* mask = (const unsigned char*)d_in[1];
    const float* w_qkv        = (const float*)d_in[2];
    const float* b_qkv        = (const float*)d_in[3];
    const float* w1           = (const float*)d_in[4];
    const float* b1           = (const float*)d_in[5];
    const float* w2           = (const float*)d_in[6];
    const float* b2           = (const float*)d_in[7];
    const float* gamma        = (const float*)d_in[8];
    const float* beta         = (const float*)d_in[9];
    float* out = (float*)d_out;
    char*  ws  = (char*)d_ws;

    unsigned short* Xbf   = (unsigned short*)(ws + 0);
    unsigned short* WqkvT = (unsigned short*)(ws + 8388608);
    unsigned short* QKVbf = (unsigned short*)(ws + 14680064);
    unsigned short* W1T   = (unsigned short*)(ws + 39845888);
    unsigned short* W2T   = (unsigned short*)(ws + 48234496);
    unsigned short* Attnbf= (unsigned short*)(ws + 56623104);
    unsigned short* Hbf   = (unsigned short*)(ws + 65011712);
    float*          F2out = (float*)(ws + 98566144);

    // prep: activations -> bf16; weights -> transposed bf16 (K-contiguous)
    cvt_f32_bf16<<<4096, 256, 0, stream>>>(emb, Xbf, 4096 * 1024);
    cvt_transpose_bf16<<<dim3(3072 / 32, 1024 / 32), 256, 0, stream>>>(
        w_qkv, WqkvT, 1024, 3072);
    cvt_transpose_bf16<<<dim3(4096 / 32, 1024 / 32), 256, 0, stream>>>(
        w1, W1T, 1024, 4096);
    cvt_transpose_bf16<<<dim3(1024 / 32, 4096 / 32), 256, 0, stream>>>(
        w2, W2T, 4096, 1024);

    // QKV projection: [4096,1024] @ [1024,3072] -> bf16
    gemm_bf16<0, 1><<<dim3(3072 / 64, 4096 / 128), 256, 0, stream>>>(
        Xbf, WqkvT, b_qkv, QKVbf, 4096, 3072, 1024);

    // flash attention: grid (S/128, B*NHEAD)
    attention_kernel<<<dim3(16, 32), 256, 0, stream>>>(QKVbf, mask, Attnbf);

    // FFN1: relu(attn @ w1 + b1) -> bf16 [4096,4096]
    gemm_bf16<1, 1><<<dim3(4096 / 64, 4096 / 128), 256, 0, stream>>>(
        Attnbf, W1T, b1, Hbf, 4096, 4096, 1024);

    // FFN2: h @ w2 + b2 -> fp32 [4096,1024]
    gemm_bf16<0, 0><<<dim3(1024 / 64, 4096 / 128), 256, 0, stream>>>(
        Hbf, W2T, b2, F2out, 4096, 1024, 4096);

    // LayerNorm -> d_out
    layernorm_kernel<<<4096, 256, 0, stream>>>(F2out, gamma, beta, out);
}
